// ParaLIF_43087111913763
// MI455X (gfx1250) — matrix operand order repaired and checked by reading, not compiled, not run on hardware
//
#include <hip/hip_runtime.h>
#include <hip/hip_bf16.h>

// ---- problem dims (fixed by the reference) ----
#define BATCH   32
#define TSTEPS  1024
#define DIN     512
#define DHID    512
#define MTOT    (BATCH * TSTEPS)          // 32768

typedef __attribute__((ext_vector_type(16))) __bf16 v16bf;
typedef __attribute__((ext_vector_type(8)))  float  v8f;

union FragAB { v16bf v; uint4 q[2]; };
union FragC  { v8f   v; float f[8]; };

// ---------------- f32 -> bf16 (round-to-nearest-even) ----------------
__device__ __forceinline__ unsigned int f2bf_pack(float a, float b) {
    union { float f; unsigned int u; } ua, ub;
    ua.f = a; ub.f = b;
    unsigned int ra = (ua.u + 0x7FFFu + ((ua.u >> 16) & 1u)) >> 16;
    unsigned int rb = (ub.u + 0x7FFFu + ((ub.u >> 16) & 1u)) >> 16;
    return (ra & 0xFFFFu) | (rb << 16);
}

__global__ void cvt_f32_bf16(const float* __restrict__ src,
                             unsigned int* __restrict__ dst, int n4) {
    int i = blockIdx.x * blockDim.x + threadIdx.x;   // one thread = 4 elements
    if (i >= n4) return;
    float4 f = reinterpret_cast<const float4*>(src)[i];
    uint2 p;
    p.x = f2bf_pack(f.x, f.y);
    p.y = f2bf_pack(f.z, f.w);
    reinterpret_cast<uint2*>(dst)[i] = p;
}

// ---------------- bf16 WMMA GEMM: X[m][n] = sum_k A[m][k] * W[n][k] ----------------
// Workgroup = 8 waves (256 thr). Each workgroup owns a 64-column N-slab whose
// full-depth B operand (64 x 512 bf16 = 64 KB) is staged once into LDS in
// WMMA fragment order; each wave then computes a 32(M) x 64(N) tile (2x4
// accumulators) reading B from LDS and A from global (bf16, L2-resident).
//
// LDS layout: 64 fragment-blocks of 1 KB, block f = ks*4 + ni (ks = K-step of
// 32, ni = 16-col group). Within a block: bytes [0,512) = per-lane q0 (16 B,
// lane-contiguous), bytes [512,1024) = per-lane q1. 16-B lane-contiguous reads
// sweep all 64 LDS banks per 16-lane phase -> conflict-free ds_load_b128.
__global__ __launch_bounds__(256) void gemm_bf16_wmma(
    const __hip_bfloat16* __restrict__ A,   // MTOT x DIN, row-major
    const __hip_bfloat16* __restrict__ W,   // DHID x DIN, row-major (B cols)
    float* __restrict__ X)                  // MTOT x DHID
{
    __shared__ uint4 ldsB[4096];            // 64 KB

    const int tid  = threadIdx.x;
    const int lane = tid & 31;              // wave32 lane, EXEC all ones
    const int wv   = tid >> 5;              // wave id 0..7
    const int hl   = lane >> 4;             // lane half (0 or 1)
    const int l16  = lane & 15;

    const int n0 = blockIdx.x * 64;
    const int m0 = blockIdx.y * 256 + wv * 32;

    // ---- stage B slab into LDS (fragment order), all 256 threads ----
#pragma unroll
    for (int i = 0; i < 8; ++i) {
        const int slot = tid + i * 256;     // 0..2047 lane-slots of 32 B
        const int l    = slot & 31;
        const int ni   = (slot >> 5) & 3;
        const int ks   = slot >> 7;         // 0..15
        const int col  = n0 + ni * 16 + (l & 15);
        const int kb   = ks * 32 + ((l >> 4) ? 8 : 0);
        const uint4* src =
            reinterpret_cast<const uint4*>(W + (size_t)col * DIN + kb);
        const int f = ks * 4 + ni;          // fragment block
        ldsB[f * 64 + l]      = src[0];     // q0 half
        ldsB[f * 64 + 32 + l] = src[2];     // q1 half
    }
    __syncthreads();

    FragC c[2][4];
#pragma unroll
    for (int mi = 0; mi < 2; ++mi)
#pragma unroll
        for (int ni = 0; ni < 4; ++ni)
#pragma unroll
            for (int r = 0; r < 8; ++r) c[mi][ni].f[r] = 0.0f;

    // A fragment rows: 16-bit A 16x32 layout — lane(0..15): M=lane, K chunks
    // [k0+8h .. +8) and [k0+16+8h .. +8) where h = lane>>4.
    const __hip_bfloat16* arow0 = A + (size_t)(m0 + l16) * DIN;
    const __hip_bfloat16* arow1 = arow0 + (size_t)16 * DIN;

#pragma unroll
    for (int ks = 0; ks < 16; ++ks) {
        const int k0 = ks * 32;

        // prefetch next K-step of the A stream (global_prefetch_b8)
        if (k0 + 32 < DIN) {
            __builtin_prefetch(arow0 + k0 + 32, 0, 3);
            __builtin_prefetch(arow1 + k0 + 32, 0, 3);
        }

        FragAB a[2], b[4];
        const int ka = k0 + hl * 8;
        a[0].q[0] = *reinterpret_cast<const uint4*>(arow0 + ka);
        a[0].q[1] = *reinterpret_cast<const uint4*>(arow0 + ka + 16);
        a[1].q[0] = *reinterpret_cast<const uint4*>(arow1 + ka);
        a[1].q[1] = *reinterpret_cast<const uint4*>(arow1 + ka + 16);

#pragma unroll
        for (int ni = 0; ni < 4; ++ni) {
            const int f = ks * 4 + ni;
            b[ni].q[0] = ldsB[f * 64 + lane];
            b[ni].q[1] = ldsB[f * 64 + 32 + lane];
        }

#pragma unroll
        for (int mi = 0; mi < 2; ++mi)
#pragma unroll
            for (int ni = 0; ni < 4; ++ni)
                c[mi][ni].v = __builtin_amdgcn_wmma_f32_16x16x32_bf16(
                    /*neg_a=*/false, a[mi].v,
                    /*neg_b=*/false, b[ni].v,
                    /*c_mod=*/(short)0, c[mi][ni].v,
                    /*reuse_a=*/false, /*reuse_b=*/false);
    }

    // Store: C/D 16x16 f32 layout — VGPR r, lanes 0..15 => M=r, lanes 16..31 => M=8+r.
#pragma unroll
    for (int mi = 0; mi < 2; ++mi) {
        const int rowbase = m0 + mi * 16 + (hl ? 8 : 0);
#pragma unroll
        for (int r = 0; r < 8; ++r) {
            float* xp = X + (size_t)(rowbase + r) * DHID + n0 + l16;
#pragma unroll
            for (int ni = 0; ni < 4; ++ni) xp[ni * 16] = c[mi][ni].f[r];
        }
    }
}

// ---------------- double-exponential IIR scan + threshold ----------------
// mem_pot = conv(X + b, c) with c[t] = (1-beta)*(t+1)*beta^t  (alpha==beta==e^-1),
// which is exactly: syn = a*syn + x ; mem = b*mem + (1-b)*syn.
__global__ void lif_scan(const float* __restrict__ X,
                         const float* __restrict__ bias,
                         const float* __restrict__ thr,
                         float* __restrict__ out)
{
    const int idx = blockIdx.x * blockDim.x + threadIdx.x;   // 0 .. BATCH*DHID-1
    const int bb = idx / DHID;
    const int h  = idx % DHID;

    const float alpha = 0.36787944117144233f;   // exp(-1)
    const float beta  = 0.36787944117144233f;
    const float omb   = 1.0f - beta;

    const float th = thr[0];
    const float bh = bias[h];

    float syn = 0.0f, mem = 0.0f;
    size_t base = (size_t)bb * TSTEPS * DHID + h;
    for (int t = 0; t < TSTEPS; ++t) {
        float x = X[base] + bh;
        syn = alpha * syn + x;
        mem = beta * mem + omb * syn;
        out[base] = (mem - th > 0.0f) ? 1.0f : 0.0f;
        base += DHID;
    }
}

// ---------------- launcher ----------------
extern "C" void kernel_launch(void* const* d_in, const int* in_sizes, int n_in,
                              void* d_out, int out_size, void* d_ws, size_t ws_size,
                              hipStream_t stream) {
    const float* inputs = (const float*)d_in[0];   // (32,1024,512) f32
    const float* W      = (const float*)d_in[1];   // (512,512) f32
    const float* bias   = (const float*)d_in[2];   // (512,) f32
    const float* thr    = (const float*)d_in[3];   // scalar f32

    char* ws = (char*)d_ws;
    const size_t A_BYTES = (size_t)MTOT * DIN * 2;       // 33,554,432
    const size_t W_BYTES = (size_t)DHID * DIN * 2;       //    524,288
    __hip_bfloat16* Abf = (__hip_bfloat16*)ws;
    __hip_bfloat16* Wbf = (__hip_bfloat16*)(ws + A_BYTES);
    float*          X   = (float*)(ws + A_BYTES + W_BYTES);   // 67,108,864 bytes

    // 1) convert activations and weights to bf16
    {
        int n4 = (MTOT * DIN) / 4;
        cvt_f32_bf16<<<(n4 + 255) / 256, 256, 0, stream>>>(inputs, (unsigned int*)Abf, n4);
        int w4 = (DHID * DIN) / 4;
        cvt_f32_bf16<<<(w4 + 255) / 256, 256, 0, stream>>>(W, (unsigned int*)Wbf, w4);
    }

    // 2) WMMA GEMM: X = A @ W^T   (grid: N/64 x M/256, 8 waves per block)
    gemm_bf16_wmma<<<dim3(DHID / 64, MTOT / 256), 256, 0, stream>>>(Abf, Wbf, X);

    // 3) IIR scan + bias + surrogate-spike threshold
    lif_scan<<<(BATCH * DHID) / 256, 256, 0, stream>>>(X, bias, thr, (float*)d_out);
}
